// MANIQA_36953898614934
// MI455X (gfx1250) — compile-verified, matrix operand order checked by reading, not run on hardware
//
#include <hip/hip_runtime.h>
#include <hip/hip_bf16.h>

typedef __attribute__((ext_vector_type(16))) __bf16 v16bf;
typedef __attribute__((ext_vector_type(8)))  float  v8f;

#define BATCH 64
#define LTOK  196
#define CDIM  512
#define NHH   4
#define HDD   128
#define KPAD  224                      // 196 padded to multiple of 32
#define MROWS (BATCH*LTOK)             // 12544
#define QELEMS (BATCH*NHH*LTOK*HDD)    // 6422528 elements per Q (and K)
#define VTOFF  (2*QELEMS)              // Vt starts after Q and K

// ---------------------------------------------------------------------------
// Fragment load per ISA 7.12.2 (16-bit A/B layout, wave32):
// lane L: r = L&15 (row of A / col of B), kb = (L>>4)*8,
// halves [0..7] = K{kb..kb+7}, halves [8..15] = K{kb+16..kb+23}.
// With [row, K] (A) or [N, K] (Bt) storage both chunks are contiguous 16B.
// ---------------------------------------------------------------------------
static __device__ __forceinline__ v16bf ldfrag(const __bf16* p) {
  v16bf f;
  uint4* fu = reinterpret_cast<uint4*>(&f);
  fu[0] = *reinterpret_cast<const uint4*>(p);
  fu[1] = *reinterpret_cast<const uint4*>(p + 16);
  return f;
}

#define WMMA_BF16(A_, B_, C_) \
  __builtin_amdgcn_wmma_f32_16x16x32_bf16(false, (A_), false, (B_), (short)0, (C_), false, false)

// ---------------------------------------------------------------------------
// Generic bf16 WMMA GEMM:  Out = A[M,K] * Bt[N,K]^T  (+ fused epilogue)
// Wave tile: 32 rows x 64 cols (8 wmma / k-step); block: 8 waves (4M x 2N)
// = 128x128 tile. blockIdx.z = batch; strides sA/sBt in elements.
// Out-of-range rows/cols are index-CLAMPED on load (never predicated): WMMA
// output (m,n) depends only on A row m / B col n, and stores are guarded, so
// clamped lanes are harmless. K must be a multiple of 32 (padded upstream).
// Modes:
//  0: pooled-proj scatter:   x[(m/49)*196 + tokOff + m%49, n] = acc + bias[n]
//  1: qkv split: Q(scaled)/K -> [b,h,l,d] bf16 ; V -> transposed [b,h,d,l(pad224)]
//  2: scores:  S[z, m, n] = acc  (fp32)
//  3: AV out:  attn[(b*196+m)*512 + h*128 + n] = bf16(acc)
//  4: bf16 out [M,N] with bias + ReLU (MLP1)
//  5: fp32 in-place residual: Out[m*512+n] += acc + bias[n]
// ---------------------------------------------------------------------------
__global__ __launch_bounds__(256) void gemm_wmma_bf16(
    const __bf16* __restrict__ A, long long sA, int lda, int M,
    const __bf16* __restrict__ Bt, long long sBt, int ldb, int N,
    int K,
    float* __restrict__ OutF, __bf16* __restrict__ OutH,
    const float* __restrict__ bias,
    int mode, int tokOff)
{
  const int z    = blockIdx.z;
  const __bf16* Ab = A  + (size_t)z * (size_t)sA;
  const __bf16* Bb = Bt + (size_t)z * (size_t)sBt;
  const int lane = threadIdx.x & 31;
  const int wave = threadIdx.x >> 5;
  const int row0 = blockIdx.y * 128 + (wave & 3) * 32;
  const int col0 = blockIdx.x * 128 + (wave >> 2) * 64;
  const int r    = lane & 15;
  const int kb   = (lane >> 4) << 3;

  // clamped per-lane source rows/cols (branch-free inner loop)
  int ra0 = row0 + r;       if (ra0 > M - 1) ra0 = M - 1;
  int ra1 = row0 + 16 + r;  if (ra1 > M - 1) ra1 = M - 1;
  int cb0 = col0 +  0 + r;  if (cb0 > N - 1) cb0 = N - 1;
  int cb1 = col0 + 16 + r;  if (cb1 > N - 1) cb1 = N - 1;
  int cb2 = col0 + 32 + r;  if (cb2 > N - 1) cb2 = N - 1;
  int cb3 = col0 + 48 + r;  if (cb3 > N - 1) cb3 = N - 1;
  const __bf16* pa0 = Ab + (size_t)ra0 * lda + kb;
  const __bf16* pa1 = Ab + (size_t)ra1 * lda + kb;
  const __bf16* pb0 = Bb + (size_t)cb0 * ldb + kb;
  const __bf16* pb1 = Bb + (size_t)cb1 * ldb + kb;
  const __bf16* pb2 = Bb + (size_t)cb2 * ldb + kb;
  const __bf16* pb3 = Bb + (size_t)cb3 * ldb + kb;

  v8f c00 = {}, c01 = {}, c02 = {}, c03 = {};
  v8f c10 = {}, c11 = {}, c12 = {}, c13 = {};

  // software-pipelined: load k-step ahead, then 8 independent wmma
  v16bf a0 = ldfrag(pa0), a1 = ldfrag(pa1);
  v16bf b0 = ldfrag(pb0), b1 = ldfrag(pb1), b2 = ldfrag(pb2), b3 = ldfrag(pb3);
  for (int k = 32; k < K; k += 32) {
    v16bf na0 = ldfrag(pa0 + k), na1 = ldfrag(pa1 + k);
    v16bf nb0 = ldfrag(pb0 + k), nb1 = ldfrag(pb1 + k);
    v16bf nb2 = ldfrag(pb2 + k), nb3 = ldfrag(pb3 + k);
    c00 = WMMA_BF16(a0, b0, c00);  c01 = WMMA_BF16(a0, b1, c01);
    c02 = WMMA_BF16(a0, b2, c02);  c03 = WMMA_BF16(a0, b3, c03);
    c10 = WMMA_BF16(a1, b0, c10);  c11 = WMMA_BF16(a1, b1, c11);
    c12 = WMMA_BF16(a1, b2, c12);  c13 = WMMA_BF16(a1, b3, c13);
    a0 = na0; a1 = na1; b0 = nb0; b1 = nb1; b2 = nb2; b3 = nb3;
  }
  c00 = WMMA_BF16(a0, b0, c00);  c01 = WMMA_BF16(a0, b1, c01);
  c02 = WMMA_BF16(a0, b2, c02);  c03 = WMMA_BF16(a0, b3, c03);
  c10 = WMMA_BF16(a1, b0, c10);  c11 = WMMA_BF16(a1, b1, c11);
  c12 = WMMA_BF16(a1, b2, c12);  c13 = WMMA_BF16(a1, b3, c13);

  v8f accs[2][4] = {{c00, c01, c02, c03}, {c10, c11, c12, c13}};
  #pragma unroll
  for (int mi = 0; mi < 2; ++mi) {
    const int rbase = row0 + mi * 16 + ((lane >> 4) << 3);  // lanes 16-31 -> +8
    #pragma unroll
    for (int ni = 0; ni < 4; ++ni) {
      const int coln = col0 + 16 * ni + r;                  // C/D: N = lane&15
      if (coln >= N) continue;
      #pragma unroll
      for (int v = 0; v < 8; ++v) {
        const int rown = rbase + v;
        if (rown >= M) continue;
        const float val = accs[mi][ni][v];
        if (mode == 0) {
          const float vv = val + bias[coln];
          const int bi = rown / 49, pp = rown - bi * 49;
          OutF[((size_t)(bi * LTOK + tokOff + pp)) * CDIM + coln] = vv;
        } else if (mode == 1) {
          const float vv = val + bias[coln];
          const int bb = rown / LTOK, l = rown - bb * LTOK;
          const int part = coln >> 9, hn = coln & 511, h = hn >> 7, d = hn & 127;
          const size_t bh = (size_t)(bb * NHH + h);
          if (part == 0)      OutH[(bh * LTOK + l) * HDD + d] = (__bf16)(vv * 0.08838834764831845f);
          else if (part == 1) OutH[QELEMS + (bh * LTOK + l) * HDD + d] = (__bf16)vv;
          else                OutH[VTOFF + (bh * HDD + d) * KPAD + l] = (__bf16)vv;
        } else if (mode == 2) {
          OutF[(size_t)z * LTOK * LTOK + (size_t)rown * LTOK + coln] = val;
        } else if (mode == 3) {
          const int bb = z >> 2, h = z & 3;
          OutH[((size_t)(bb * LTOK + rown)) * CDIM + h * HDD + coln] = (__bf16)val;
        } else if (mode == 4) {
          const float vv = val + bias[coln];
          OutH[(size_t)rown * N + coln] = (__bf16)(vv > 0.f ? vv : 0.f);
        } else {
          const size_t a5 = (size_t)rown * CDIM + coln;
          OutF[a5] += val + bias[coln];
        }
      }
    }
  }
}

// ---------------- small support kernels ----------------

// fp32 [NL,K,N] -> bf16 [NL,N,K] (weight transpose for B-operand layout)
__global__ void k_transpose_w(const float* __restrict__ in, __bf16* __restrict__ out,
                              int NL, int K, int N) {
  int idx = blockIdx.x * 256 + threadIdx.x;
  int total = NL * N * K;
  if (idx >= total) return;
  int l = idx / (N * K), rem = idx - l * N * K;
  int n = rem / K, k = rem - n * K;
  out[idx] = (__bf16)in[(size_t)l * K * N + (size_t)k * N + n];
}

__global__ void k_convert(const float* __restrict__ in, __bf16* __restrict__ out, int n) {
  int idx = blockIdx.x * 256 + threadIdx.x;
  if (idx < n) out[idx] = (__bf16)in[idx];
}

__global__ void k_zero_bf16(__bf16* __restrict__ p, int n) {
  int idx = blockIdx.x * 256 + threadIdx.x;
  if (idx < n) p[idx] = (__bf16)0.0f;
}

// adaptive 7x7 avg pool -> bf16 token matrix Apool[b*49+p, c]
__global__ void k_pool(const float* __restrict__ f, __bf16* __restrict__ Apool,
                       int Cin, int H, int fdiv) {
  int idx = blockIdx.x * 256 + threadIdx.x;
  int total = BATCH * Cin * 49;
  if (idx >= total) return;
  int b = idx / (Cin * 49), rem = idx - b * Cin * 49;
  int c = rem / 49, p = rem - c * 49;
  int ph = p / 7, pw = p - ph * 7;
  const float* base = f + (((size_t)b * Cin + c) * H + ph * fdiv) * H + pw * fdiv;
  float s = 0.f;
  for (int i = 0; i < fdiv; ++i)
    for (int j = 0; j < fdiv; ++j) s += base[i * H + j];
  Apool[((size_t)b * 49 + p) * Cin + c] = (__bf16)(s / (float)(fdiv * fdiv));
}

// LayerNorm over 512 ch, fp32 in -> bf16 out; one wave per row
__global__ __launch_bounds__(256) void k_ln(const float* __restrict__ x,
                                            const float* __restrict__ g,
                                            const float* __restrict__ b,
                                            __bf16* __restrict__ out) {
  int row  = blockIdx.x * 8 + (threadIdx.x >> 5);
  int lane = threadIdx.x & 31;
  const float* xr = x + (size_t)row * CDIM;
  float vals[16], s = 0.f, s2 = 0.f;
  #pragma unroll
  for (int i = 0; i < 16; ++i) {
    float v = xr[lane + i * 32];
    vals[i] = v; s += v; s2 += v * v;
  }
  for (int o = 16; o > 0; o >>= 1) { s += __shfl_xor(s, o, 32); s2 += __shfl_xor(s2, o, 32); }
  float mean = s * (1.f / CDIM);
  float var  = s2 * (1.f / CDIM) - mean * mean;
  float rstd = rsqrtf(var + 1e-5f);
  __bf16* orow = out + (size_t)row * CDIM;
  #pragma unroll
  for (int i = 0; i < 16; ++i) {
    int c = lane + i * 32;
    orow[c] = (__bf16)((vals[i] - mean) * rstd * g[c] + b[c]);
  }
}

// relative-position bias matrix: biasM[h,i,j] = bt[((dy+13)*27 + dx+13)*4 + h]
__global__ void k_biasmat(const float* __restrict__ bt, float* __restrict__ biasM) {
  int idx = blockIdx.x * 256 + threadIdx.x;
  int total = NHH * LTOK * LTOK;
  if (idx >= total) return;
  int h = idx / (LTOK * LTOK), rem = idx - h * LTOK * LTOK;
  int i = rem / LTOK, j = rem - i * LTOK;
  int dy = i / 14 - j / 14 + 13;
  int dx = i % 14 - j % 14 + 13;
  biasM[idx] = bt[(dy * 27 + dx) * NHH + h];
}

// softmax over 196 cols (+bias), writes bf16 P padded to 224 cols; wave per row
__global__ __launch_bounds__(256) void k_softmax(const float* __restrict__ S,
                                                 const float* __restrict__ biasM,
                                                 __bf16* __restrict__ P) {
  int rid  = blockIdx.x * 8 + (threadIdx.x >> 5);  // b*4*196 rows
  int lane = threadIdx.x & 31;
  int zz = rid / LTOK, i = rid - zz * LTOK, h = zz & 3;
  const float* srow = S + (size_t)zz * LTOK * LTOK + (size_t)i * LTOK;
  const float* brow = biasM + ((size_t)h * LTOK + i) * LTOK;
  float v[7], mx = -1e30f;
  #pragma unroll
  for (int t = 0; t < 7; ++t) {
    int j = t * 32 + lane;
    float val = (j < LTOK) ? srow[j] + brow[j] : -1e30f;
    v[t] = val; mx = fmaxf(mx, val);
  }
  for (int o = 16; o > 0; o >>= 1) mx = fmaxf(mx, __shfl_xor(mx, o, 32));
  float sum = 0.f;
  #pragma unroll
  for (int t = 0; t < 7; ++t) { v[t] = __expf(v[t] - mx); sum += v[t]; }
  for (int o = 16; o > 0; o >>= 1) sum += __shfl_xor(sum, o, 32);
  float inv = 1.f / sum;
  __bf16* prow = P + (size_t)zz * LTOK * KPAD + (size_t)i * KPAD;
  #pragma unroll
  for (int t = 0; t < 7; ++t) {
    int j = t * 32 + lane;                      // covers exactly 0..223
    prow[j] = (__bf16)((j < LTOK) ? v[t] * inv : 0.f);
  }
}

// final mean over 196 tokens -> out[b, c] fp32
__global__ void k_mean(const float* __restrict__ x, float* __restrict__ out) {
  int idx = blockIdx.x * 256 + threadIdx.x;
  if (idx >= BATCH * CDIM) return;
  int b = idx / CDIM, c = idx - b * CDIM;
  float s = 0.f;
  for (int l = 0; l < LTOK; ++l) s += x[((size_t)b * LTOK + l) * CDIM + c];
  out[idx] = s * (1.f / LTOK);
}

// ---------------- host orchestration ----------------
static inline int cdiv(int a, int b) { return (a + b - 1) / b; }

extern "C" void kernel_launch(void* const* d_in, const int* in_sizes, int n_in,
                              void* d_out, int out_size, void* d_ws, size_t ws_size,
                              hipStream_t stream) {
  (void)in_sizes; (void)n_in; (void)out_size; (void)ws_size;
  const float* feat[4]  = {(const float*)d_in[0], (const float*)d_in[1],
                           (const float*)d_in[2], (const float*)d_in[3]};
  const float* projw[4] = {(const float*)d_in[4], (const float*)d_in[6],
                           (const float*)d_in[8], (const float*)d_in[10]};
  const float* projb[4] = {(const float*)d_in[5], (const float*)d_in[7],
                           (const float*)d_in[9], (const float*)d_in[11]};
  const float* ln1g = (const float*)d_in[12]; const float* ln1b = (const float*)d_in[13];
  const float* qkvw = (const float*)d_in[14]; const float* qkvb = (const float*)d_in[15];
  const float* btab = (const float*)d_in[16];
  const float* apw  = (const float*)d_in[17]; const float* apb = (const float*)d_in[18];
  const float* ln2g = (const float*)d_in[19]; const float* ln2b = (const float*)d_in[20];
  const float* w1   = (const float*)d_in[21]; const float* b1  = (const float*)d_in[22];
  const float* w2   = (const float*)d_in[23]; const float* b2  = (const float*)d_in[24];

  char* wsp = (char*)d_ws;
  size_t off = 0;
  auto alloc = [&](size_t bytes) -> void* {
    void* p = wsp + off;
    off = (off + bytes + 255) & ~(size_t)255;
    return p;
  };

  const int cins[4] = {96, 192, 384, 768};
  const int cpre[4] = {0, 96, 288, 672};
  const int hs[4]   = {56, 28, 14, 7};

  __bf16* qkvT   = (__bf16*)alloc((size_t)2 * 1536 * 512 * 2);
  __bf16* pwT    = (__bf16*)alloc((size_t)2 * 512 * 512 * 2);
  __bf16* w1T    = (__bf16*)alloc((size_t)2 * 2048 * 512 * 2);
  __bf16* w2T    = (__bf16*)alloc((size_t)2 * 512 * 2048 * 2);
  __bf16* projwb = (__bf16*)alloc((size_t)512 * 1440 * 2);
  __bf16* Apool  = (__bf16*)alloc((size_t)3136 * 1440 * 2);
  float*  x      = (float*)alloc((size_t)MROWS * CDIM * 4);
  __bf16* xn     = (__bf16*)alloc((size_t)MROWS * CDIM * 2);
  // attention region (H1 for the MLP aliases its head once attention is done)
  const size_t vtBytes = (size_t)BATCH * NHH * HDD * KPAD * 2;
  const size_t sBytes  = (size_t)BATCH * NHH * LTOK * LTOK * 4;
  const size_t pBytes  = (size_t)BATCH * NHH * LTOK * KPAD * 2;
  char* region = (char*)alloc((size_t)2 * QELEMS * 2 + vtBytes + sBytes + pBytes);
  __bf16* Qb = (__bf16*)region;                         // K at Qb+QELEMS, Vt at Qb+VTOFF
  __bf16* Vt = Qb + VTOFF;
  float*  S  = (float*)(region + (size_t)2 * QELEMS * 2 + vtBytes);
  __bf16* P  = (__bf16*)((char*)S + sBytes);
  __bf16* H1 = (__bf16*)region;                         // aliases Q/K/Vt (dead by MLP time)
  __bf16* attn  = (__bf16*)alloc((size_t)MROWS * CDIM * 2);
  float*  biasM = (float*)alloc((size_t)NHH * LTOK * LTOK * 4);

  // ---- weight conversion ----
  k_transpose_w<<<cdiv(2 * 1536 * 512, 256), 256, 0, stream>>>(qkvw, qkvT, 2, 512, 1536);
  k_transpose_w<<<cdiv(2 * 512 * 512, 256), 256, 0, stream>>>(apw, pwT, 2, 512, 512);
  k_transpose_w<<<cdiv(2 * 2048 * 512, 256), 256, 0, stream>>>(w1, w1T, 2, 512, 2048);
  k_transpose_w<<<cdiv(2 * 512 * 2048, 256), 256, 0, stream>>>(w2, w2T, 2, 2048, 512);
  for (int s = 0; s < 4; ++s)
    k_convert<<<cdiv(512 * cins[s], 256), 256, 0, stream>>>(
        projw[s], projwb + (size_t)512 * cpre[s], 512 * cins[s]);

  // ---- pooling + projection into token matrix x [B,196,512] ----
  for (int s = 0; s < 4; ++s)
    k_pool<<<cdiv(BATCH * cins[s] * 49, 256), 256, 0, stream>>>(
        feat[s], Apool + (size_t)3136 * cpre[s], cins[s], hs[s], hs[s] / 7);
  for (int s = 0; s < 4; ++s)
    gemm_wmma_bf16<<<dim3(cdiv(512, 128), cdiv(3136, 128), 1), 256, 0, stream>>>(
        Apool + (size_t)3136 * cpre[s], 0, cins[s], 3136,
        projwb + (size_t)512 * cpre[s], 0, cins[s], 512, cins[s],
        x, nullptr, projb[s], /*mode=*/0, /*tokOff=*/49 * s);

  // ---- transformer layers ----
  for (int l = 0; l < 2; ++l) {
    k_biasmat<<<cdiv(NHH * LTOK * LTOK, 256), 256, 0, stream>>>(btab + (size_t)l * 729 * NHH, biasM);
    k_ln<<<MROWS / 8, 256, 0, stream>>>(x, ln1g + l * CDIM, ln1b + l * CDIM, xn);
    k_zero_bf16<<<cdiv(BATCH * NHH * HDD * KPAD, 256), 256, 0, stream>>>(Vt, BATCH * NHH * HDD * KPAD);
    // qkv: [12544,512] x [512,1536] -> scaled Q / K / Vt (bf16)
    gemm_wmma_bf16<<<dim3(1536 / 128, MROWS / 128, 1), 256, 0, stream>>>(
        xn, 0, 512, MROWS, qkvT + (size_t)l * 1536 * 512, 0, 512, 1536, 512,
        nullptr, Qb, qkvb + (size_t)l * 1536, /*mode=*/1, 0);
    // scores: per (b,h): [196,128] x [128,196]
    gemm_wmma_bf16<<<dim3(cdiv(LTOK, 128), cdiv(LTOK, 128), BATCH * NHH), 256, 0, stream>>>(
        Qb, (long long)LTOK * HDD, HDD, LTOK,
        Qb + QELEMS, (long long)LTOK * HDD, HDD, LTOK, HDD,
        S, nullptr, nullptr, /*mode=*/2, 0);
    k_softmax<<<BATCH * NHH * LTOK / 8, 256, 0, stream>>>(S, biasM, P);
    // AV: per (b,h): [196,224] x [224,128] -> attn [B,L,C] bf16
    gemm_wmma_bf16<<<dim3(1, cdiv(LTOK, 128), BATCH * NHH), 256, 0, stream>>>(
        P, (long long)LTOK * KPAD, KPAD, LTOK,
        Vt, (long long)HDD * KPAD, KPAD, HDD, KPAD,
        nullptr, attn, nullptr, /*mode=*/3, 0);
    // output projection + residual (in-place on x)
    gemm_wmma_bf16<<<dim3(512 / 128, MROWS / 128, 1), 256, 0, stream>>>(
        attn, 0, 512, MROWS, pwT + (size_t)l * 512 * 512, 0, 512, 512, 512,
        x, nullptr, apb + (size_t)l * CDIM, /*mode=*/5, 0);
    // MLP
    k_ln<<<MROWS / 8, 256, 0, stream>>>(x, ln2g + l * CDIM, ln2b + l * CDIM, xn);
    gemm_wmma_bf16<<<dim3(2048 / 128, MROWS / 128, 1), 256, 0, stream>>>(
        xn, 0, 512, MROWS, w1T + (size_t)l * 2048 * 512, 0, 512, 2048, 512,
        nullptr, H1, b1 + (size_t)l * 2048, /*mode=*/4, 0);
    gemm_wmma_bf16<<<dim3(512 / 128, MROWS / 128, 1), 256, 0, stream>>>(
        H1, 0, 2048, MROWS, w2T + (size_t)l * 512 * 2048, 0, 2048, 512, 2048,
        x, nullptr, b2 + (size_t)l * CDIM, /*mode=*/5, 0);
  }

  k_mean<<<cdiv(BATCH * CDIM, 256), 256, 0, stream>>>(x, (float*)d_out);
}
